// HMM_9113920602174
// MI455X (gfx1250) — compile-verified, matrix operand order checked by reading, not run on hardware
//
#include <hip/hip_runtime.h>

// CDNA5 / gfx1250 HMM forward scan, single-barrier ping-pong version, 2x unrolled.
// alpha_t = log( P_{t-1} @ E ) + rowmax_{t-1} + emis_t,  P = exp(alpha - rowmax), E = exp(trans)
// 4 WGs (16 batch rows each) x 4 waves (16-col N tile each); 2x v_wmma_f32_16x16x32_f16 per step.
// A-fragments built in registers from LDS f32 state; ONE ds-only barrier per step;
// emissions double-buffered in registers with no copy (loop unrolled by 2).

typedef __attribute__((ext_vector_type(16))) _Float16 v16h;
typedef __attribute__((ext_vector_type(8)))  float    v8f;

#define HMM_B 64
#define HMM_T 4096
#define HMM_S 64
#define SPAD  69   // padded LDS row stride: conflict-free for A-reads and C-writes

#define LOG2E 1.44269504088896340736f
#define LN2   0.69314718055994530942f

static __device__ __forceinline__ float fast_exp(float x) {  // e^x, args never extreme here
    return __builtin_amdgcn_exp2f(x * LOG2E);
}

// Workgroup barrier that orders ONLY LDS traffic: global loads/stores keep draining.
static __device__ __forceinline__ void wg_barrier_dsonly() {
    asm volatile("s_wait_dscnt 0x0\n\t"
                 "s_barrier_signal -1\n\t"
                 "s_barrier_wait -1" ::: "memory");
}

__launch_bounds__(128, 1)
__global__ void hmm_forward_wmma(const float* __restrict__ trans,
                                 const float* __restrict__ emis,
                                 float* __restrict__ alpha_out,
                                 float* __restrict__ logz_out)
{
    __shared__ float s_buf[2][16][SPAD];    // ping-pong alpha state (16 batch rows x 64 states)

    const int tid  = threadIdx.x;           // 0..127
    const int wave = tid >> 5;              // 0..3 -> N tile
    const int lane = tid & 31;
    const int b0   = blockIdx.x * 16;       // batch base for this WG
    const int n0   = wave * 16;             // column base for this wave

    const int m16  = lane & 15;             // A-fragment row owned by this lane
    const int kb   = (lane >> 4) * 8;       // A-fragment K base (8-half groups)

    // ---- B fragments: E[k][n] = exp(trans[k][n]); resident in VGPRs for the whole scan ----
    v16h bfrag0, bfrag1;
    {
        const int n = n0 + m16;
        #pragma unroll
        for (int j = 0; j < 8; ++j) {
            bfrag0[j]     = (_Float16)fast_exp(trans[(kb + j)      * HMM_S + n]);
            bfrag0[j + 8] = (_Float16)fast_exp(trans[(16 + kb + j) * HMM_S + n]);
            bfrag1[j]     = (_Float16)fast_exp(trans[(32 + kb + j) * HMM_S + n]);
            bfrag1[j + 8] = (_Float16)fast_exp(trans[(48 + kb + j) * HMM_S + n]);
        }
    }

    // ---- t = 0: alpha0 = emis[:,0,:] into buffer 0 ----
    #pragma unroll
    for (int e = 0; e < 8; ++e) {
        int idx = tid * 8 + e;              // covers 16x64
        int m = idx >> 6, n = idx & 63;
        size_t gi = (size_t)(b0 + m) * (HMM_T * HMM_S) + n;
        float v = emis[gi];
        s_buf[0][m][n] = v;
        alpha_out[gi] = v;
    }
    __syncthreads();

    const int nn    = n0 + m16;             // phase-C column
    const int mbase = (lane >> 4) * 8;      // phase-C row base (C/D layout)

    // One scan step: consume evc (emissions for t), load evn (emissions for t+1, clamped).
    auto step = [&](int t, float (*rd)[SPAD], float (*wr)[SPAD],
                    const float* evc, float* evn) {
        // emission loads for t+1 (register double buffer; consumed next step)
        const size_t tn = (t + 1 < HMM_T) ? (size_t)(t + 1) : (size_t)t;
        #pragma unroll
        for (int g = 0; g < 8; ++g)
            evn[g] = emis[(size_t)(b0 + mbase + g) * (HMM_T * HMM_S) + tn * HMM_S + nn];

        // L2 prefetch ~16 steps ahead; 128 threads exactly tile the WG's 16x64 f32 block
        {
            int tp = (t + 16 < HMM_T) ? t + 16 : HMM_T - 1;
            __builtin_prefetch(&emis[(size_t)(b0 + (tid >> 3)) * (HMM_T * HMM_S)
                                     + (size_t)tp * HMM_S + (tid & 7) * 8], 0, 1);
        }

        // ---- phase A (barrier-free): load this lane's full A rows as f32, rowmax, exp ----
        float av0[8], av1[8], av2[8], av3[8];
        #pragma unroll
        for (int j = 0; j < 8; ++j) {
            av0[j] = rd[m16][kb + j];
            av1[j] = rd[m16][16 + kb + j];
            av2[j] = rd[m16][32 + kb + j];
            av3[j] = rd[m16][48 + kb + j];
        }
        // balanced max tree (depth ~7)
        float mm[8];
        #pragma unroll
        for (int j = 0; j < 8; ++j)
            mm[j] = fmaxf(fmaxf(av0[j], av1[j]), fmaxf(av2[j], av3[j]));
        float x0 = fmaxf(mm[0], mm[1]), x1 = fmaxf(mm[2], mm[3]);
        float x2 = fmaxf(mm[4], mm[5]), x3 = fmaxf(mm[6], mm[7]);
        float mx = fmaxf(fmaxf(x0, x1), fmaxf(x2, x3));
        mx = fmaxf(mx, __shfl_xor(mx, 16, 32));   // pair lane holds the other 32 columns

        // p = exp(av - mx) as one fma + v_exp per element
        const float nb = -mx * LOG2E;
        v16h a0, a1;
        #pragma unroll
        for (int j = 0; j < 8; ++j) {
            a0[j]     = (_Float16)__builtin_amdgcn_exp2f(fmaf(av0[j], LOG2E, nb));
            a0[j + 8] = (_Float16)__builtin_amdgcn_exp2f(fmaf(av1[j], LOG2E, nb));
            a1[j]     = (_Float16)__builtin_amdgcn_exp2f(fmaf(av2[j], LOG2E, nb));
            a1[j + 8] = (_Float16)__builtin_amdgcn_exp2f(fmaf(av3[j], LOG2E, nb));
        }

        // every wave holds all 16 row maxes (lane r and r+16 own row r): gather for phase C
        float base[8];
        #pragma unroll
        for (int g = 0; g < 8; ++g)
            base[g] = __shfl(mx, mbase + g, 32) + evc[g];

        // ---- phase B: s = P(16x64) @ E(64x16-tile), two chained WMMAs ----
        v8f c = {};
        c = __builtin_amdgcn_wmma_f32_16x16x32_f16(false, a0, false, bfrag0,
                                                   (short)0, c, false, false);
        c = __builtin_amdgcn_wmma_f32_16x16x32_f16(false, a1, false, bfrag1,
                                                   (short)0, c, false, false);

        // ---- phase C: alpha_t = ln2*log2(s) + (rowmax + emis_t); write state + stream out ----
        #pragma unroll
        for (int g = 0; g < 8; ++g) {
            int m = mbase + g;               // C/D layout: VGPR g -> M = mbase+g, N = nn
            float aval = fmaf(LN2, __builtin_amdgcn_logf(c[g]), base[g]);
            wr[m][nn] = aval;
            alpha_out[(size_t)(b0 + m) * (HMM_T * HMM_S) + (size_t)t * HMM_S + nn] = aval;
        }

        wg_barrier_dsonly();    // orders LDS only; alpha stores / emis loads keep flowing
    };

    // preload emissions for t = 1
    float evA[8], evB[8];
    #pragma unroll
    for (int g = 0; g < 8; ++g)
        evA[g] = emis[(size_t)(b0 + mbase + g) * (HMM_T * HMM_S) + (size_t)HMM_S + nn];

    // 2x-unrolled scan: odd t reads buf0/writes buf1, even t the reverse; no ev copies
    int t = 1;
    for (; t + 1 < HMM_T; t += 2) {
        step(t,     s_buf[0], s_buf[1], evA, evB);
        step(t + 1, s_buf[1], s_buf[0], evB, evA);
    }
    step(t, s_buf[0], s_buf[1], evA, evB);    // t = HMM_T-1 (odd since HMM_T is even)

    // ---- log_Z over alpha[:, T-1, :] (final state in buffer (T-1)&1 = 1) ----
    float (*fin)[SPAD] = s_buf[(HMM_T - 1) & 1];
    const int row = tid >> 3, chunk = tid & 7;   // 8 aligned lanes per row
    float mx = -3.402823466e38f;
    #pragma unroll
    for (int e = 0; e < 8; ++e) mx = fmaxf(mx, fin[row][chunk * 8 + e]);
    mx = fmaxf(mx, __shfl_xor(mx, 1, 8));
    mx = fmaxf(mx, __shfl_xor(mx, 2, 8));
    mx = fmaxf(mx, __shfl_xor(mx, 4, 8));
    float sm = 0.f;
    #pragma unroll
    for (int e = 0; e < 8; ++e) sm += fast_exp(fin[row][chunk * 8 + e] - mx);
    sm += __shfl_xor(sm, 1, 8);
    sm += __shfl_xor(sm, 2, 8);
    sm += __shfl_xor(sm, 4, 8);
    if (chunk == 0) logz_out[b0 + row] = fmaf(LN2, __builtin_amdgcn_logf(sm), mx);
}

extern "C" void kernel_launch(void* const* d_in, const int* in_sizes, int n_in,
                              void* d_out, int out_size, void* d_ws, size_t ws_size,
                              hipStream_t stream)
{
    (void)in_sizes; (void)n_in; (void)d_ws; (void)ws_size; (void)out_size;
    const float* trans = (const float*)d_in[0];   // (S,S) f32
    const float* emis  = (const float*)d_in[1];   // (B,T,S) f32
    // d_in[2] = seq_lens: unused by the reference computation.
    float* out  = (float*)d_out;                  // alpha (B*T*S) then log_Z (B)
    float* logz = out + (size_t)HMM_B * HMM_T * HMM_S;

    hmm_forward_wmma<<<dim3(HMM_B / 16), dim3(128), 0, stream>>>(trans, emis, out, logz);
}
